// MultiRNNCell_18519898980861
// MI455X (gfx1250) — matrix operand order repaired and checked by reading, not compiled, not run on hardware
//
#include <hip/hip_runtime.h>
#include <hip/hip_bf16.h>

// Problem dims (match reference)
#define Bk 64
#define Tk 128
#define Dk 256
#define Hk 1024
#define Gk (4*Hk)

typedef __attribute__((ext_vector_type(16))) __bf16 v16bf;
typedef __attribute__((ext_vector_type(8)))  float  v8f;

// ---------------------------------------------------------------------------
// WMMA fragment loaders (contiguous k = 16h+s in A and B alike, wave32)
// ---------------------------------------------------------------------------
// A-matrix 16x32 bf16: lanes 0-15 hold row M=lane, K = 0..15;
// lanes 16-31 hold row M=lane-16, K = 16..31. One 32B contiguous load.
static __device__ __forceinline__ v16bf load_a_frag(const __bf16* base, int lda, int lane) {
  const int r  = lane & 15;
  const int kh = lane >> 4;
  const __bf16* p = base + r * lda + kh * 16;
  return *reinterpret_cast<const v16bf*>(p);
}

// B-matrix 32x16 bf16 with B = W^T (W row-major [N,K]): lane holds column
// n = lane&15; lanes 0-15 carry K=0..15, lanes 16-31 carry K=16..31 --
// i.e. 16 contiguous bf16 from one weight row: a single aligned 32B load.
static __device__ __forceinline__ v16bf load_b_frag(const __bf16* Wrow0, int ldw, int lane) {
  const int c  = lane & 15;
  const int kh = lane >> 4;
  const __bf16* p = Wrow0 + c * ldw + kh * 16;
  return *reinterpret_cast<const v16bf*>(p);
}

static __device__ __forceinline__ float sigmf(float x) { return 1.0f / (1.0f + __expf(-x)); }

// ---------------------------------------------------------------------------
// One-time per launch: fp32 -> bf16 weights, fold Wih1+Whh1, fold biases, h0=0
// ---------------------------------------------------------------------------
__global__ void prep_kernel(const float* __restrict__ Wih0, const float* __restrict__ Whh0,
                            const float* __restrict__ bih0, const float* __restrict__ bhh0,
                            const float* __restrict__ Wih1, const float* __restrict__ Whh1,
                            const float* __restrict__ bih1, const float* __restrict__ bhh1,
                            const float* __restrict__ Wout,
                            __bf16* __restrict__ Wih0B, __bf16* __restrict__ Whh0B,
                            __bf16* __restrict__ Wc1B,  __bf16* __restrict__ WoutB,
                            float* __restrict__ b0, float* __restrict__ bc1,
                            float* __restrict__ hA, __bf16* __restrict__ hAB) {
  const int n = Gk * Hk;
  const int stride = gridDim.x * blockDim.x;
  for (int i = blockIdx.x * blockDim.x + threadIdx.x; i < n; i += stride) {
    Whh0B[i] = (__bf16)Whh0[i];
    Wc1B[i]  = (__bf16)(Wih1[i] + Whh1[i]);      // layer-1: x == hx, fold weights
    if (i < Gk * Dk) Wih0B[i] = (__bf16)Wih0[i];
    if (i < Dk * Hk) WoutB[i] = (__bf16)Wout[i];
    if (i < Gk) { b0[i] = bih0[i] + bhh0[i]; bc1[i] = bih1[i] + bhh1[i]; }
    if (i < Bk * Hk) { hA[i] = 0.0f; hAB[i] = (__bf16)0.0f; }
  }
}

// ---------------------------------------------------------------------------
// est = h @ Wout^T + bout ; store into output slot (t-1) ; imputed = m*x+(1-m)*est
// grid = 16 n-tiles (D=256), block = 128 (4 waves = 4 M-tiles of B=64)
// ---------------------------------------------------------------------------
__global__ __launch_bounds__(128) void est_impute_kernel(
    const __bf16* __restrict__ hAB, const __bf16* __restrict__ WoutB,
    const float* __restrict__ bout,
    const float* __restrict__ X, const float* __restrict__ Msk,
    float* __restrict__ out, __bf16* __restrict__ impB,
    int t, int do_store, int do_impute) {
  const int lane = threadIdx.x & 31;
  const int m0 = (threadIdx.x >> 5) << 4;     // M-tile (batch rows)
  const int n0 = blockIdx.x << 4;             // N-tile (feature cols of D)

  v8f acc = {};
  for (int kk = 0; kk < Hk; kk += 32) {
    v16bf a = load_a_frag(hAB + m0 * Hk + kk, Hk, lane);
    v16bf w = load_b_frag(WoutB + n0 * Hk + kk, Hk, lane);
    if (kk + 64 < Hk)
      __builtin_prefetch(WoutB + n0 * Hk + kk + 64, 0, 1);
    acc = __builtin_amdgcn_wmma_f32_16x16x32_bf16(false, a, false, w, (short)0, acc, false, false);
  }

  const int n = n0 + (lane & 15);
  const float bo = bout[n];
  const int rbase = m0 + ((lane >> 4) << 3);
#pragma unroll
  for (int e = 0; e < 8; ++e) {
    const int bi = rbase + e;                  // batch index
    const float est = acc[e] + bo;
    if (do_store) out[((size_t)bi * Tk + (t - 1)) * Dk + n] = est;
    if (do_impute) {
      const size_t idx = ((size_t)bi * Tk + t) * Dk + n;
      const float xv = X[idx];
      const float mv = Msk[idx];
      impB[bi * Dk + n] = (__bf16)(mv * xv + (1.0f - mv) * est);
    }
  }
}

// ---------------------------------------------------------------------------
// Fused LSTM-cell gate GEMM + nonlinearity, 4-way K-split per gate tile.
// Computes only i/g/o blocks (f-gate is dead: cell state is always zero).
//   gates = A1 @ W1^T [+ A2 @ W2^T] + bias ;  h = sigm(o)*tanh(sigm(i)*tanh(g))
// Block = 256 thr = 8 waves = 2 n-tile groups x 4 K-slices.
// Grid = 32 blocks -> 64 n-tiles covering H=1024; each wave does 4 M-tiles and
// a 1/4 K-chunk; partials combine via LDS ds_add_f32, epilogue runs from LDS.
// ---------------------------------------------------------------------------
__global__ __launch_bounds__(256) void gates_kernel(
    const __bf16* __restrict__ A1, int ldA1, const __bf16* __restrict__ W1, int ldW1, int K1,
    const __bf16* __restrict__ A2, int ldA2, const __bf16* __restrict__ W2, int ldW2, int K2,
    const float* __restrict__ bias,
    float* __restrict__ outF, __bf16* __restrict__ outB) {
  __shared__ float gLDS[2][3 * 64 * 16];           // 2 groups x 12 KB

  const int lane = threadIdx.x & 31;
  const int wave = threadIdx.x >> 5;               // 0..7
  const int grp  = wave >> 2;                      // n-tile group within block
  const int ks   = wave & 3;                       // K-slice index
  const int n0   = (blockIdx.x * 2 + grp) << 4;    // N-tile within H
  float* gbuf = gLDS[grp];
  const int tig = threadIdx.x & 127;               // thread id within group

  // zero the group's accumulation buffer
  for (int i = tig; i < 3 * 64 * 16; i += 128) gbuf[i] = 0.0f;
  __syncthreads();

  v8f acc[3][4];
#pragma unroll
  for (int g = 0; g < 3; ++g)
#pragma unroll
    for (int mt = 0; mt < 4; ++mt) acc[g][mt] = (v8f){0,0,0,0,0,0,0,0};

  // this wave's K-chunk in the concatenated [A1 | A2] K space (32-aligned)
  const int KT = K1 + K2;
  const int chunk = KT >> 2;
  const int lo = ks * chunk, hi = lo + chunk;

  const __bf16* Aseg = A1; int lda = ldA1;
  const __bf16* Wseg = W1; int ldw = ldW1;
  int slo = (lo < K1) ? lo : K1;
  int shi = (hi < K1) ? hi : K1;
  for (int seg = 0; seg < 2; ++seg) {
    const __bf16* wI = Wseg + (size_t)(n0)          * ldw;
    const __bf16* wG = Wseg + (size_t)(n0 + 2 * Hk) * ldw;
    const __bf16* wO = Wseg + (size_t)(n0 + 3 * Hk) * ldw;
    for (int kk = slo; kk < shi; kk += 32) {
      v16bf a[4];
#pragma unroll
      for (int mt = 0; mt < 4; ++mt)
        a[mt] = load_a_frag(Aseg + (size_t)(mt << 4) * lda + kk, lda, lane);
      v16bf bI = load_b_frag(wI + kk, ldw, lane);
      v16bf bG = load_b_frag(wG + kk, ldw, lane);
      v16bf bO = load_b_frag(wO + kk, ldw, lane);
      if (kk + 64 < shi) {
        __builtin_prefetch(wI + kk + 64, 0, 1);
        __builtin_prefetch(wG + kk + 64, 0, 1);
        __builtin_prefetch(wO + kk + 64, 0, 1);
      }
#pragma unroll
      for (int mt = 0; mt < 4; ++mt) {
        acc[0][mt] = __builtin_amdgcn_wmma_f32_16x16x32_bf16(false, a[mt], false, bI, (short)0, acc[0][mt], false, false);
        acc[1][mt] = __builtin_amdgcn_wmma_f32_16x16x32_bf16(false, a[mt], false, bG, (short)0, acc[1][mt], false, false);
        acc[2][mt] = __builtin_amdgcn_wmma_f32_16x16x32_bf16(false, a[mt], false, bO, (short)0, acc[2][mt], false, false);
      }
    }
    if (seg == 0) {
      if (K2 <= 0) break;
      // segment-2 portion of this wave's chunk (local K coordinates)
      slo = ((lo > K1) ? lo : K1) - K1;
      shi = ((hi > K1) ? hi : K1) - K1;
      if (slo >= shi) break;
      Aseg = A2; lda = ldA2; Wseg = W2; ldw = ldW2;
    }
  }

  // combine the 4 K-slice partials in LDS (ds_add_f32)
  const int col = lane & 15;
  const int rb  = (lane >> 4) << 3;
#pragma unroll
  for (int g = 0; g < 3; ++g)
#pragma unroll
    for (int mt = 0; mt < 4; ++mt)
#pragma unroll
      for (int e = 0; e < 8; ++e) {
        const int row = (mt << 4) + rb + e;
        atomicAdd(&gbuf[(g * 64 + row) * 16 + col], acc[g][mt][e]);
      }
  __syncthreads();

  // cooperative fused LSTM epilogue (c_in == 0 -> c = sigm(i)*tanh(g))
  for (int i = tig; i < 64 * 16; i += 128) {
    const int row = i >> 4;
    const int c   = i & 15;
    const int n   = n0 + c;
    const float gi = sigmf(gbuf[(0 * 64 + row) * 16 + c] + bias[n]);
    const float gg = tanhf(gbuf[(1 * 64 + row) * 16 + c] + bias[2 * Hk + n]);
    const float go = sigmf(gbuf[(2 * 64 + row) * 16 + c] + bias[3 * Hk + n]);
    const float hv = go * tanhf(gi * gg);
    outB[(size_t)row * Hk + n] = (__bf16)hv;
    if (outF) outF[(size_t)row * Hk + n] = hv;
  }
}

// ---------------------------------------------------------------------------
extern "C" void kernel_launch(void* const* d_in, const int* in_sizes, int n_in,
                              void* d_out, int out_size, void* d_ws, size_t ws_size,
                              hipStream_t stream) {
  (void)in_sizes; (void)n_in; (void)out_size; (void)ws_size;
  const float* X    = (const float*)d_in[0];
  const float* Msk  = (const float*)d_in[1];
  const float* Wih0 = (const float*)d_in[2];
  const float* Whh0 = (const float*)d_in[3];
  const float* bih0 = (const float*)d_in[4];
  const float* bhh0 = (const float*)d_in[5];
  const float* Wih1 = (const float*)d_in[6];
  const float* Whh1 = (const float*)d_in[7];
  const float* bih1 = (const float*)d_in[8];
  const float* bhh1 = (const float*)d_in[9];
  const float* Wout = (const float*)d_in[10];
  const float* bout = (const float*)d_in[11];
  float* out = (float*)d_out;

  // workspace carve-up (256B aligned), ~19.1 MiB total
  char* w = (char*)d_ws;
  auto alloc = [&](size_t bytes) {
    char* p = w; w += (bytes + 255) & ~(size_t)255; return p;
  };
  __bf16* Whh0B = (__bf16*)alloc((size_t)Gk * Hk * 2);
  __bf16* Wc1B  = (__bf16*)alloc((size_t)Gk * Hk * 2);
  __bf16* Wih0B = (__bf16*)alloc((size_t)Gk * Dk * 2);
  __bf16* WoutB = (__bf16*)alloc((size_t)Dk * Hk * 2);
  float*  b0    = (float*) alloc((size_t)Gk * 4);
  float*  bc1   = (float*) alloc((size_t)Gk * 4);
  float*  hA    = (float*) alloc((size_t)Bk * Hk * 4);   // h carried across steps (fp32)
  __bf16* hAB   = (__bf16*)alloc((size_t)Bk * Hk * 2);   // h carried across steps (bf16)
  __bf16* hBB   = (__bf16*)alloc((size_t)Bk * Hk * 2);   // layer-0 output (bf16)
  __bf16* impB  = (__bf16*)alloc((size_t)Bk * Dk * 2);   // imputed input (bf16)

  prep_kernel<<<512, 256, 0, stream>>>(Wih0, Whh0, bih0, bhh0, Wih1, Whh1, bih1, bhh1,
                                       Wout, Wih0B, Whh0B, Wc1B, WoutB, b0, bc1, hA, hAB);

  for (int t = 0; t < Tk; ++t) {
    // est from h_t -> out slot (t-1); imputed input for this step
    est_impute_kernel<<<16, 128, 0, stream>>>(hAB, WoutB, bout, X, Msk, out, impB,
                                              t, (t >= 1) ? 1 : 0, 1);
    // layer 0: gates = imputed@Wih0^T + h@Whh0^T (+b) -> hB
    gates_kernel<<<32, 256, 0, stream>>>(impB, Dk, Wih0B, Dk, Dk,
                                         hAB,  Hk, Whh0B, Hk, Hk,
                                         b0, nullptr, hBB);
    // layer 1: gates = hB@(Wih1+Whh1)^T (+b) -> hA (fp32 + bf16)
    gates_kernel<<<32, 256, 0, stream>>>(hBB, Hk, Wc1B, Hk, Hk,
                                         nullptr, 0, nullptr, 0, 0,
                                         bc1, hA, hAB);
  }
  // final estimation from h_T -> out slot T-1
  est_impute_kernel<<<16, 128, 0, stream>>>(hAB, WoutB, bout, X, Msk, out, impB, Tk, 1, 0);
  // h_final (fp32) appended after [B,T,D] block
  hipMemcpyAsync(out + (size_t)Bk * Tk * Dk, hA, (size_t)Bk * Hk * sizeof(float),
                 hipMemcpyDeviceToDevice, stream);
}
